// Nets_9337258902417
// MI455X (gfx1250) — matrix operand order repaired and checked by reading, not compiled, not run on hardware
//
#include <hip/hip_runtime.h>
#include <hip/hip_bf16.h>

typedef __bf16 bf16_t;
typedef __attribute__((ext_vector_type(16))) __bf16 v16bf;
typedef __attribute__((ext_vector_type(8)))  __bf16 v8bf;
typedef __attribute__((ext_vector_type(8)))  float  v8f;

#define INPUT_DIM 20000
#define UNITS     4096
#define NNZ_CNT   2000000
#define BATCH     2048

// ---------------- kernel 1: zero fp32 dense-kernel scratch ----------------
__global__ void k_zero_f32(float4* __restrict__ p, long n4) {
  long i = (long)blockIdx.x * blockDim.x + threadIdx.x;
  if (i < n4) p[i] = make_float4(0.f, 0.f, 0.f, 0.f);
}

// ------------- kernel 2: scatter nnz -> dense fp32 [INPUT_DIM][UNITS] -----
__global__ void k_scatter(const float* __restrict__ vals,
                          const int* __restrict__ ind,
                          float* __restrict__ Kd) {
  int i = blockIdx.x * blockDim.x + threadIdx.x;
  if (i < NNZ_CNT) {
    int r = ind[2 * i];
    int c = ind[2 * i + 1];
    atomicAdd(&Kd[(long)r * UNITS + c], vals[i]);
  }
}

// ------------- kernel 3: convert inputs fp32 -> bf16 ([BATCH][K]) ---------
__global__ void k_cvt_bf16(const float4* __restrict__ in, v8bf* __restrict__ out,
                           long n8) {
  long i = (long)blockIdx.x * blockDim.x + threadIdx.x;
  if (i < n8) {
    float4 a = in[2 * i];
    float4 b = in[2 * i + 1];
    v8bf o = {(__bf16)a.x, (__bf16)a.y, (__bf16)a.z, (__bf16)a.w,
              (__bf16)b.x, (__bf16)b.y, (__bf16)b.z, (__bf16)b.w};
    out[i] = o;
  }
}

// --- kernel 4: transpose+convert dense kernel fp32[K][N] -> bf16[N][K] ----
__global__ void k_transpose_cvt(const float* __restrict__ Kd,
                                bf16_t* __restrict__ Bt) {
  __shared__ float tile[32][33];
  int tx = threadIdx.x & 31;
  int ty = threadIdx.x >> 5;            // 0..7
  long k0 = (long)blockIdx.x * 32;      // along INPUT_DIM
  long n0 = (long)blockIdx.y * 32;      // along UNITS
#pragma unroll
  for (int r = 0; r < 32; r += 8)
    tile[ty + r][tx] = Kd[(k0 + ty + r) * UNITS + (n0 + tx)];
  __syncthreads();
#pragma unroll
  for (int r = 0; r < 32; r += 8)
    Bt[(n0 + ty + r) * INPUT_DIM + (k0 + tx)] = (__bf16)tile[tx][ty + r];
}

// ---------------- kernel 5: WMMA bf16 GEMM + bias + ELU -------------------
#define BM 128
#define BN 128
#define BK 160              // 5 x K32 WMMA steps per LDS stage; 20000 % 160 == 0
#define LDT 168             // padded LDS row stride in elements (336B, 16B aligned)

static __device__ __forceinline__ v16bf join8(v8bf lo, v8bf hi) {
  return __builtin_shufflevector(lo, hi, 0, 1, 2, 3, 4, 5, 6, 7,
                                 8, 9, 10, 11, 12, 13, 14, 15);
}

// CDNA5 async copy: global -> LDS, no VGPR staging, tracked by ASYNCcnt.
// VDST operand carries the LDS byte address (flat-pointer low 32 bits,
// inside the LDS aperture).
static __device__ __forceinline__ void async_g2l_b128(const bf16_t* gptr,
                                                      bf16_t* lptr) {
  unsigned lds_addr = (unsigned)(unsigned long long)lptr;
  asm volatile("global_load_async_to_lds_b128 %0, %1, off"
               :
               : "v"(lds_addr), "v"(gptr)
               : "memory");
}

// Issue one BMxBK A-tile + BNxBK B-tile into LDS (20 async b128 per thread).
static __device__ __forceinline__ void stage_tile_async(
    const bf16_t* __restrict__ A, const bf16_t* __restrict__ B,
    bf16_t* sAbuf, bf16_t* sBbuf, long rowBase, long colBase, int kk,
    int tid) {
#pragma unroll
  for (int i = 0; i < 10; ++i) {
    int idx = tid + 256 * i;  // 0..2559 ; 20 uint4-chunks per 160-elem row
    int row = idx / 20;
    int q   = idx % 20;
    async_g2l_b128(A + (rowBase + row) * INPUT_DIM + kk + q * 8,
                   sAbuf + row * LDT + q * 8);
    async_g2l_b128(B + (colBase + row) * INPUT_DIM + kk + q * 8,
                   sBbuf + row * LDT + q * 8);
  }
}

__global__ __launch_bounds__(256) void k_gemm_bias_elu(
    const bf16_t* __restrict__ A,    // [BATCH][INPUT_DIM] bf16
    const bf16_t* __restrict__ B,    // [UNITS][INPUT_DIM] bf16 (pre-transposed)
    const float* __restrict__ bias,  // [UNITS]
    float* __restrict__ out) {       // [BATCH][UNITS]
  __shared__ __align__(16) bf16_t sA[2][BM * LDT];  // double buffered: 172KB total
  __shared__ __align__(16) bf16_t sB[2][BN * LDT];

  const int tid  = threadIdx.x;
  const int lane = tid & 31;
  const int wave = tid >> 5;
  const int wm = wave & 3;   // 4 wave-rows of 32 output rows
  const int wn = wave >> 2;  // 2 wave-cols of 64 output cols
  const long rowBase = (long)blockIdx.x * BM;
  const long colBase = (long)blockIdx.y * BN;

  const int ml = lane & 15;  // matrix row/col within 16
  const int kh = lane >> 4;  // K half-select per ISA 16-bit fragment layout

  v8f acc[2][4] = {};

  constexpr int NT = INPUT_DIM / BK;  // 125 tiles

  // Prologue: prefetch tile 0 into buffer 0.
  stage_tile_async(A, B, sA[0], sB[0], rowBase, colBase, 0, tid);

  for (int t = 0; t < NT; ++t) {
    const int cur = t & 1;

    // Prefetch next tile into the other buffer (its previous readers passed
    // the trailing barrier of iteration t-1, so it is free).
    if (t + 1 < NT) {
      stage_tile_async(A, B, sA[cur ^ 1], sB[cur ^ 1], rowBase, colBase,
                       (t + 1) * BK, tid);
      // <=20 outstanding -> current tile's (in-order) async loads completed.
      asm volatile("s_wait_asynccnt 0x14" ::: "memory");
    } else {
      asm volatile("s_wait_asynccnt 0x0" ::: "memory");
    }
    __syncthreads();

    const bf16_t* __restrict__ cA = sA[cur];
    const bf16_t* __restrict__ cB = sB[cur];
#pragma unroll
    for (int s = 0; s < BK / 32; ++s) {
      const int ko = s * 32;
      v16bf af[2], bfr[4];
      // A fragment 16x32: lane<16 holds K 0-7 & 16-23; lane>=16 holds 8-15 & 24-31
#pragma unroll
      for (int tm = 0; tm < 2; ++tm) {
        const bf16_t* p = cA + (wm * 32 + tm * 16 + ml) * LDT + ko + kh * 8;
        v8bf lo = *(const v8bf*)p;
        v8bf hi = *(const v8bf*)(p + 16);
        af[tm] = join8(lo, hi);
      }
      // B fragment 32x16: lane<16 holds K 0-7 & 16-23 of column n; lane>=16 holds 8-15 & 24-31
#pragma unroll
      for (int tn = 0; tn < 4; ++tn) {
        const bf16_t* p = cB + (wn * 64 + tn * 16 + ml) * LDT + ko + kh * 8;
        v8bf lo = *(const v8bf*)p;
        v8bf hi = *(const v8bf*)(p + 16);
        bfr[tn] = join8(lo, hi);
      }
#pragma unroll
      for (int tm = 0; tm < 2; ++tm)
#pragma unroll
        for (int tn = 0; tn < 4; ++tn)
          acc[tm][tn] = __builtin_amdgcn_wmma_f32_16x16x32_bf16(
              false, af[tm], false, bfr[tn], (short)0, acc[tm][tn], false,
              false);
    }
    __syncthreads();  // all waves done reading buffer `cur` before reuse
  }

  // Epilogue. C/D layout: VGPR j -> row (j + 8*kh), col = ml.
#pragma unroll
  for (int tn = 0; tn < 4; ++tn) {
    long col = colBase + wn * 64 + tn * 16 + ml;
    float bv = bias[col];
#pragma unroll
    for (int tm = 0; tm < 2; ++tm) {
      long r0 = rowBase + wm * 32 + tm * 16 + kh * 8;
#pragma unroll
      for (int j = 0; j < 8; ++j) {
        float x = acc[tm][tn][j] + bv;
        out[(r0 + j) * UNITS + col] = x > 0.f ? x : __expf(x) - 1.f;
      }
    }
  }
}

// --------------------------------------------------------------------------
extern "C" void kernel_launch(void* const* d_in, const int* in_sizes, int n_in,
                              void* d_out, int out_size, void* d_ws,
                              size_t ws_size, hipStream_t stream) {
  (void)in_sizes; (void)n_in; (void)out_size; (void)ws_size;

  const float* inputs = (const float*)d_in[0];      // [2048*20000] f32
  const float* vals   = (const float*)d_in[1];      // [2000000] f32
  const float* bias   = (const float*)d_in[2];      // [4096] f32
  const int*   ind    = (const int*)d_in[3];        // [2000000*2] i32
  float* out = (float*)d_out;

  char* ws = (char*)d_ws;
  const long nK = (long)INPUT_DIM * UNITS;          // 81,920,000
  float*  K32 = (float*)ws;                                       // 327.7 MB
  bf16_t* Bt  = (bf16_t*)(ws + (size_t)nK * 4);                   // 163.8 MB
  bf16_t* Ap  = (bf16_t*)(ws + (size_t)nK * 4 + (size_t)nK * 2);  // 81.9 MB

  // 1) zero dense kernel scratch (every call: deterministic, replay-safe)
  long n4 = nK / 4;  // 20,480,000 float4
  k_zero_f32<<<(unsigned)(n4 / 256), 256, 0, stream>>>((float4*)K32, n4);

  // 2) scatter-add nnz
  k_scatter<<<(NNZ_CNT + 255) / 256, 256, 0, stream>>>(vals, ind, K32);

  // 3) inputs fp32 -> bf16
  long n8 = (long)BATCH * INPUT_DIM / 8;  // 5,120,000
  k_cvt_bf16<<<(unsigned)(n8 / 256), 256, 0, stream>>>((const float4*)inputs,
                                                       (v8bf*)Ap, n8);

  // 4) dense kernel fp32[K][N] -> bf16[N][K]
  k_transpose_cvt<<<dim3(INPUT_DIM / 32, UNITS / 32), 256, 0, stream>>>(K32, Bt);

  // 5) WMMA GEMM + bias + ELU (double-buffered async-LDS pipeline)
  k_gemm_bias_elu<<<dim3(BATCH / BM, UNITS / BN), 256, 0, stream>>>(Ap, Bt, bias,
                                                                    out);
}
